// Atten_Head_48455821034028
// MI455X (gfx1250) — matrix operand order repaired and checked by reading, not compiled, not run on hardware
//
#include <hip/hip_runtime.h>

// Problem constants (match reference)
#define NB    8
#define SEQ   2048
#define DIN   1024
#define DOUT  1024
#define MROWS (NB * SEQ)          // 16384
#define SCALE 0.03125f            // 1/sqrt(1024)

typedef __attribute__((ext_vector_type(16))) _Float16 v16h;
typedef __attribute__((ext_vector_type(8)))  _Float16 v8h;
typedef __attribute__((ext_vector_type(8)))  float    v8f;
typedef unsigned int u32;
typedef __attribute__((ext_vector_type(4))) u32 u32x4;
typedef __attribute__((ext_vector_type(8))) u32 u32x8;

__device__ __forceinline__ v8f wmma16(v16h a, v16h b, v8f c) {
    // D = A(16x32 f16) * B(32x16 f16) + C(16x16 f32)
    return __builtin_amdgcn_wmma_f32_16x16x32_f16(
        /*neg_a=*/false, a, /*neg_b=*/false, b,
        /*c_mod=*/(short)0, c, /*reuse_a=*/false, /*reuse_b=*/false);
}

// Build A-fragment (16x32, row-major source):
// lane half h: elements 0-15 hold K = 16h..16h+15 (contiguous, matches B)
__device__ __forceinline__ v16h load_afrag(const _Float16* row_ptr, int kbase, int h) {
    return *(const v16h*)(row_ptr + kbase + h * 16);
}

__device__ __forceinline__ v8f zero_v8f() {
    v8f z = {0.f, 0.f, 0.f, 0.f, 0.f, 0.f, 0.f, 0.f};
    return z;
}

// ---------------------------------------------------------------------------
// Kernel 1: W (fp32 [DIN][DOUT]) -> WT (f16 [DOUT][DIN]) via LDS tile transpose
// grid: (DOUT/32, DIN/32, 3), block: 256
// ---------------------------------------------------------------------------
__global__ __launch_bounds__(256) void wt_transpose_kernel(
    const float* __restrict__ wk, const float* __restrict__ wv,
    const float* __restrict__ wq, _Float16* __restrict__ wt)
{
    const int z = blockIdx.z;
    const float* W = (z == 0) ? wk : ((z == 1) ? wv : wq);
    _Float16* O = wt + (size_t)z * DIN * DOUT;

    __shared__ float tile[32][33];
    const int o0 = blockIdx.x * 32;
    const int i0 = blockIdx.y * 32;
    const int tx = threadIdx.x & 31;
    const int ty = threadIdx.x >> 5;     // 0..7

#pragma unroll
    for (int e = 0; e < 4; ++e)
        tile[ty + 8 * e][tx] = W[(size_t)(i0 + ty + 8 * e) * DOUT + o0 + tx];
    __syncthreads();
#pragma unroll
    for (int e = 0; e < 4; ++e)
        O[(size_t)(o0 + ty + 8 * e) * DIN + i0 + tx] = (_Float16)tile[tx][ty + 8 * e];
}

// ---------------------------------------------------------------------------
// Kernel 2: QKV projection GEMM.  X fp32 [MROWS][DIN] * W -> f16 [MROWS][DOUT]
// z=0: K (row-major), z=1: V stored transposed [B][DOUT][SEQ], z=2: Q (row-major)
// Block tile 64(M) x 128(N), 8 waves of 16x64, K-chunks of 32 through LDS.
// B tile staged by the Tensor Data Mover (tensor_load_to_lds, TENSORcnt).
// grid: (DOUT/128, MROWS/64, 3), block: 256
// ---------------------------------------------------------------------------
__global__ __launch_bounds__(256, 1) void qkv_proj_kernel(
    const float* __restrict__ xk, const float* __restrict__ xv,
    const float* __restrict__ xq, const _Float16* __restrict__ wt,
    _Float16* __restrict__ kb, _Float16* __restrict__ vtb,
    _Float16* __restrict__ qb)
{
    const int z = blockIdx.z;
    const float*    X = (z == 0) ? xk : ((z == 1) ? xv : xq);
    const _Float16* W = wt + (size_t)z * DIN * DOUT;   // transposed: [DOUT][DIN]

    const int n0 = blockIdx.x * 128;
    const int m0 = blockIdx.y * 64;

    __shared__ __align__(32) _Float16 As[64 * 32];
    __shared__ __align__(32) _Float16 Bs[128 * 32];

    const int tid  = threadIdx.x;
    const int lane = tid & 31;
    const int wid  = tid >> 5;
    const int wm   = wid & 3;        // 4 M sub-tiles of 16
    const int wn   = wid >> 2;       // 2 N sub-tiles of 64
    const int h    = lane >> 4;      // lane half
    const int ln   = lane & 15;

    // TDM descriptor group 1 (invariant): tile 32(k) x 128(n) of 2-byte elems
    // from a [DOUT][DIN] tensor with row stride DIN, packed row-major into Bs.
    u32x8 g1;
    g1[0] = (1u << 16);              // data_size = 1 -> 2 bytes
    g1[1] = ((u32)DIN) << 16;        // tensor_dim0[15:0] (bits 79:48 field)
    g1[2] = ((u32)DOUT) << 16;       // tensor_dim0 hi = 0 | tensor_dim1[15:0]
    g1[3] = (32u << 16);             // tensor_dim1 hi = 0 | tile_dim0 = 32
    g1[4] = 128u;                    // tile_dim1 = 128, tile_dim2 = 0
    g1[5] = (u32)DIN;                // tensor_dim0_stride[31:0] = 1024
    g1[6] = 0u;                      // stride0 hi | tensor_dim1_stride lo
    g1[7] = 0u;
    const u32 lds_b = (u32)(size_t)(void*)Bs;

    v8f acc[4];
#pragma unroll
    for (int s = 0; s < 4; ++s) acc[s] = zero_v8f();

    for (int k0 = 0; k0 < DIN; k0 += 32) {
        // ---- B tile via Tensor Data Mover (one DMA per workgroup) ----
        if (wid == 0) {
            const unsigned long long ga =
                (unsigned long long)(const void*)(W + (size_t)n0 * DIN + k0);
            u32x4 g0;
            g0[0] = 1u;                            // count = 1 (valid D#)
            g0[1] = lds_b;                         // LDS dest byte address
            g0[2] = (u32)ga;                       // global_addr[31:0]
            g0[3] = (u32)(ga >> 32) | (2u << 30);  // global_addr[56:32]|type=2
            asm volatile("tensor_load_to_lds %0, %1"
                         :: "s"(g0), "s"(g1) : "memory");
        }

        // ---- A tile: 64x32 fp32 -> f16 in LDS (512 float4, 2 per thread) ----
#pragma unroll
        for (int e = 0; e < 2; ++e) {
            const int lin = tid + e * 256;
            const int r   = lin >> 3;
            const int c4  = lin & 7;
            const float4 f = *(const float4*)(X + (size_t)(m0 + r) * DIN + k0 + c4 * 4);
            _Float16* dst = As + r * 32 + c4 * 4;
            dst[0] = (_Float16)f.x; dst[1] = (_Float16)f.y;
            dst[2] = (_Float16)f.z; dst[3] = (_Float16)f.w;
        }

        __builtin_amdgcn_s_wait_tensorcnt((short)0);
        __syncthreads();

        const v16h a = load_afrag(As + (wm * 16 + ln) * 32, 0, h);
#pragma unroll
        for (int s = 0; s < 4; ++s) {
            // B-frag: lane = col n, 16 contiguous K at 16h
            const v16h b = *(const v16h*)(Bs + (wn * 64 + s * 16 + ln) * 32 + h * 16);
            acc[s] = wmma16(a, b, acc[s]);
        }
        __syncthreads();
    }

    // Store f16 results.  C layout: VGPR r -> row r (lanes 0-15) / 8+r (16-31)
#pragma unroll
    for (int s = 0; s < 4; ++s) {
#pragma unroll
        for (int r = 0; r < 8; ++r) {
            const int row = m0 + wm * 16 + r + 8 * h;
            const int col = n0 + wn * 64 + s * 16 + ln;
            const _Float16 v = (_Float16)acc[s][r];
            if (z == 1) {
                const int bb = row / SEQ;
                const int sl = row - bb * SEQ;
                vtb[((size_t)bb * DOUT + col) * SEQ + sl] = v;   // V^T [B][D][S]
            } else if (z == 0) {
                kb[(size_t)row * DOUT + col] = v;
            } else {
                qb[(size_t)row * DOUT + col] = v;
            }
        }
    }
}

// ---------------------------------------------------------------------------
// Kernel 3: causal flash attention, single pass, online softmax.
// Workgroup: 32 query rows, key tiles of 64, 8 waves.
//   score phase: wave (wqs = wid&1, wk = wid>>1) -> 16x16 score tile, full-D WMMA
//   PV phase:    wave (wqs, wd = wid>>1) -> 16 q-rows x 256 d-cols, 16 f32 accums
// grid: (SEQ/32, NB), block: 256
// ---------------------------------------------------------------------------
__global__ __launch_bounds__(256, 1) void flash_attn_kernel(
    const _Float16* __restrict__ qb, const _Float16* __restrict__ kb,
    const _Float16* __restrict__ vtb, float* __restrict__ out)
{
    const int b  = blockIdx.y;
    const int q0 = blockIdx.x * 32;

    const int tid  = threadIdx.x;
    const int lane = tid & 31;
    const int wid  = tid >> 5;
    const int h    = lane >> 4;
    const int ln   = lane & 15;
    const int wqs  = wid & 1;        // q sub-tile (16 rows)
    const int wk   = wid >> 1;       // score: key sub-tile / PV: d quarter

    __shared__ __align__(32) float    Ss[32 * 64];
    __shared__ __align__(32) _Float16 Ps[32 * 64];
    __shared__ float Ms[32], Ls[32], Al[32];

    if (tid < 32) { Ms[tid] = -__builtin_inff(); Ls[tid] = 0.f; }

    v8f o_acc[16];
#pragma unroll
    for (int t = 0; t < 16; ++t) o_acc[t] = zero_v8f();

    const _Float16* Qrow  = qb  + (size_t)(b * SEQ + q0 + wqs * 16 + ln) * DIN;
    const _Float16* Kbase = kb  + (size_t)b * SEQ * DIN;
    const _Float16* Vb    = vtb + (size_t)b * DOUT * SEQ;

    const int qmax = q0 + 31;
    const int nkt  = qmax / 64 + 1;

    __syncthreads();

    for (int kt = 0; kt < nkt; ++kt) {
        const int k0t = kt * 64;

        // ---- score phase: S = Q K^T over full D ----
        v8f sacc = zero_v8f();
        const _Float16* Krow = Kbase + (size_t)(k0t + wk * 16 + ln) * DIN;
        if (kt + 1 < nkt)
            __builtin_prefetch(Kbase + (size_t)(k0t + 64 + wk * 16 + ln) * DIN, 0, 1);
        for (int d = 0; d < DIN; d += 32) {
            const v16h a = load_afrag(Qrow, d, h);
            const v16h kf = *(const v16h*)(Krow + d + h * 16);
            sacc = wmma16(a, kf, sacc);
        }
#pragma unroll
        for (int r = 0; r < 8; ++r)
            Ss[(wqs * 16 + r + 8 * h) * 64 + wk * 16 + ln] = sacc[r];
        __syncthreads();

        // ---- online softmax: 32 rows x 64 cols, 8 lanes per row ----
        {
            const int row   = tid >> 3;
            const int j     = tid & 7;
            const int q_idx = q0 + row;
            const float mold = Ms[row];
            const float lold = Ls[row];
            float v[8];
            float mloc = -__builtin_inff();
#pragma unroll
            for (int c = 0; c < 8; ++c) {
                const int key = k0t + j * 8 + c;
                const float s = Ss[row * 64 + j * 8 + c] * SCALE;
                v[c] = (key <= q_idx) ? s : -__builtin_inff();
                mloc = fmaxf(mloc, v[c]);
            }
#pragma unroll
            for (int off = 1; off < 8; off <<= 1)
                mloc = fmaxf(mloc, __shfl_xor(mloc, off, 32));
            const float mnew = fmaxf(mold, mloc);
            float sum = 0.f;
#pragma unroll
            for (int c = 0; c < 8; ++c) {
                const float p = __expf(v[c] - mnew);
                sum += p;
                Ps[row * 64 + j * 8 + c] = (_Float16)p;
            }
#pragma unroll
            for (int off = 1; off < 8; off <<= 1)
                sum += __shfl_xor(sum, off, 32);
            if (j == 0) {
                const float alpha = __expf(mold - mnew);
                Ms[row] = mnew;
                Ls[row] = lold * alpha + sum;
                Al[row] = alpha;
            }
        }
        __syncthreads();

        // ---- rescale running output, then PV WMMAs ----
        float alr[8];
#pragma unroll
        for (int r = 0; r < 8; ++r) alr[r] = Al[wqs * 16 + r + 8 * h];
#pragma unroll
        for (int t = 0; t < 16; ++t)
#pragma unroll
            for (int r = 0; r < 8; ++r) o_acc[t][r] *= alr[r];

#pragma unroll
        for (int kc = 0; kc < 2; ++kc) {
            const v16h a = load_afrag(Ps + (wqs * 16 + ln) * 64, kc * 32, h);
#pragma unroll
            for (int t = 0; t < 16; ++t) {
                const int dc = wk * 256 + t * 16 + ln;
                const v16h vf = *(const v16h*)(Vb + (size_t)dc * SEQ + k0t + kc * 32 + h * 16);
                o_acc[t] = wmma16(a, vf, o_acc[t]);
            }
        }
        // next-iter barrier #1 orders Ps reads vs. next softmax writes
    }

    // ---- epilogue: normalize by l and store fp32 ----
    float invl[8];
#pragma unroll
    for (int r = 0; r < 8; ++r) invl[r] = 1.0f / Ls[wqs * 16 + r + 8 * h];
#pragma unroll
    for (int t = 0; t < 16; ++t) {
#pragma unroll
        for (int r = 0; r < 8; ++r) {
            const int row = q0 + wqs * 16 + r + 8 * h;
            const int col = wk * 256 + t * 16 + ln;
            out[((size_t)(b * SEQ) + row) * DOUT + col] = o_acc[t][r] * invl[r];
        }
    }
}

// ---------------------------------------------------------------------------
extern "C" void kernel_launch(void* const* d_in, const int* in_sizes, int n_in,
                              void* d_out, int out_size, void* d_ws, size_t ws_size,
                              hipStream_t stream) {
    (void)in_sizes; (void)n_in; (void)out_size; (void)ws_size;

    const float* xk  = (const float*)d_in[0];
    const float* xv  = (const float*)d_in[1];
    const float* xq  = (const float*)d_in[2];
    const float* wkw = (const float*)d_in[3];
    const float* wvw = (const float*)d_in[4];
    const float* wqw = (const float*)d_in[5];

    char* ws = (char*)d_ws;
    const size_t qkv_bytes = (size_t)MROWS * DOUT * sizeof(_Float16);   // 32 MB each
    _Float16* kb  = (_Float16*)(ws);
    _Float16* vtb = (_Float16*)(ws + qkv_bytes);
    _Float16* qb  = (_Float16*)(ws + 2 * qkv_bytes);
    _Float16* wt  = (_Float16*)(ws + 3 * qkv_bytes);                    // 3 x 2 MB

    wt_transpose_kernel<<<dim3(DOUT / 32, DIN / 32, 3), 256, 0, stream>>>(
        wkw, wvw, wqw, wt);
    qkv_proj_kernel<<<dim3(DOUT / 128, MROWS / 64, 3), 256, 0, stream>>>(
        xk, xv, xq, wt, kb, vtb, qb);
    flash_attn_kernel<<<dim3(SEQ / 32, NB), 256, 0, stream>>>(
        qb, kb, vtb, (float*)d_out);
}
